// THEGCNSamplerModel_10479720202342
// MI455X (gfx1250) — matrix operand order repaired and checked by reading, not compiled, not run on hardware
//
#include <hip/hip_runtime.h>
#include <hip/hip_bf16.h>

// ---------------- model constants (from reference) ----------------
#define N_SUB    100000
#define N_EDGES  1000000
#define IN_F     64
#define HID      128
#define TIMEF    64
#define BATCH    10000
#define BN_EPS   1e-5f
#define WV       4            // waves per block (blockDim = 128, wave32)

typedef __attribute__((ext_vector_type(16))) _Float16 v16h;
typedef __attribute__((ext_vector_type(8)))  float    v8f;

union Frag { v16h v; float4 f[2]; };   // 32 bytes

__device__ __forceinline__ float tanh_fast(float x) {
  x = fminf(fmaxf(x, -15.f), 15.f);
  float e = __expf(2.f * x);
  return (e - 1.f) / (e + 1.f);
}

// all device pointers, passed by value
struct P {
  // inputs
  const float* x_all; const int* node_ids; const int* e_src; const int* e_dst;
  const float* e_dt;
  const float* clfW1; const float* clfW2; const float* clfW3;
  const float* clfb1; const float* clfb2; const float* clfb3;
  const float* bn1b; const float* bn1g; const float* bn1m; const float* bn1v;
  const float* bn2b; const float* bn2g; const float* bn2m; const float* bn2v;
  const float* freq; const float* phase;
  const float* projW; const float* projb;
  const float* s0W1; const float* s0W2; const float* s0b1; const float* s0b2;
  const float* s0bnb; const float* s0bng; const float* s0bnm; const float* s0bnv;
  const float* s1W1; const float* s1W2; const float* s1b1; const float* s1b2;
  const float* s1bnb; const float* s1bng; const float* s1bnm; const float* s1bnv;
  const float* tW1; const float* tW2; const float* tb1; const float* tb2;
  // workspace
  float* x; _Float16* x_h; float* h; _Float16* h_h;
  float* aggA; float* aggB; float* cntA; float* cntB;
  _Float16* tW1t; _Float16* tW2t; _Float16* projWt;
  _Float16* s0W1t; _Float16* s0W2t; _Float16* s1W1t; _Float16* s1W2t;
  _Float16* cW1t; _Float16* cW2t;
  float* bnScale; float* bnShift;   // [s0:128][s1:128][bn1:128][bn2:64] (bn1/bn2 have clf bias folded in)
  float* out;
};

// -------- K0: weight transpose->f16 + BN folding (clf biases folded into shifts) --------
__global__ void k_prep(P p) {
  int t = blockIdx.x * blockDim.x + threadIdx.x;
  int nt = gridDim.x * blockDim.x;
  for (int i = t; i < 96 * 192; i += nt) { int n = i / 192, k = i % 192; p.tW1t[i] = (_Float16)p.tW1[k * 96 + n]; }
  for (int i = t; i < 64 * 96;  i += nt) { int n = i / 96,  k = i % 96;  p.tW2t[i] = (_Float16)p.tW2[k * 64 + n]; }
  for (int i = t; i < 128 * 64; i += nt) { int n = i / 64,  k = i % 64;  p.projWt[i] = (_Float16)p.projW[k * 128 + n]; }
  for (int i = t; i < 128 * 256; i += nt) {
    int n = i / 256, k = i % 256;
    p.s0W1t[i] = (_Float16)p.s0W1[k * 128 + n];
    p.s1W1t[i] = (_Float16)p.s1W1[k * 128 + n];
  }
  for (int i = t; i < 128 * 128; i += nt) {
    int n = i / 128, k = i % 128;
    p.s0W2t[i] = (_Float16)p.s0W2[k * 128 + n];
    p.s1W2t[i] = (_Float16)p.s1W2[k * 128 + n];
    p.cW1t[i]  = (_Float16)p.clfW1[k * 128 + n];
  }
  for (int i = t; i < 64 * 128; i += nt) { int n = i / 128, k = i % 128; p.cW2t[i] = (_Float16)p.clfW2[k * 64 + n]; }
  for (int i = t; i < 128; i += nt) {
    float sc;
    sc = p.s0bng[i] * rsqrtf(p.s0bnv[i] + BN_EPS); p.bnScale[i] = sc;       p.bnShift[i] = p.s0bnb[i] - p.s0bnm[i] * sc;
    sc = p.s1bng[i] * rsqrtf(p.s1bnv[i] + BN_EPS); p.bnScale[128 + i] = sc; p.bnShift[128 + i] = p.s1bnb[i] - p.s1bnm[i] * sc;
    // classifier bn1: fold gemm bias b1 into the shift:  (acc + b1)*sc + (beta - mean*sc)
    sc = p.bn1g[i]  * rsqrtf(p.bn1v[i]  + BN_EPS);
    p.bnScale[256 + i] = sc;
    p.bnShift[256 + i] = p.bn1b[i] + (p.clfb1[i] - p.bn1m[i]) * sc;
  }
  for (int i = t; i < 64; i += nt) {
    float sc = p.bn2g[i] * rsqrtf(p.bn2v[i] + BN_EPS);
    p.bnScale[384 + i] = sc;
    p.bnShift[384 + i] = p.bn2b[i] + (p.clfb2[i] - p.bn2m[i]) * sc;
  }
}

// -------- K1: gather x = x_all[node_ids]; zero aggA/cntA --------
__global__ void k_gather(P p) {
  long t = blockIdx.x * (long)blockDim.x + threadIdx.x;
  long nt = (long)gridDim.x * blockDim.x;
  for (long i = t; i < (long)N_SUB * IN_F; i += nt) {
    int node = (int)(i >> 6);
    int nid = p.node_ids[node];
    float v = p.x_all[(long)nid * IN_F + (i & 63)];
    p.x[i] = v; p.x_h[i] = (_Float16)v; p.aggA[i] = 0.f;
  }
  for (long i = t; i < N_SUB; i += nt) p.cntA[i] = 0.f;
}

// -------- K2: phase-1 temporal edge MLP (m=192 -> 96 -> 64) --------
__global__ __launch_bounds__(128, 1) void k_edge1(P p) {
  __shared__ _Float16 mS[WV][16 * 200];   // 192 + pad 8
  __shared__ _Float16 h1S[WV][16 * 104];  // 96 + pad 8
  const int lane = threadIdx.x & 31, wid = threadIdx.x >> 5;
  _Float16* mW = mS[wid]; _Float16* h1W = h1S[wid];
  int w = blockIdx.x * WV + wid, nw = gridDim.x * WV;
  for (int tile = w; tile < N_EDGES / 16; tile += nw) {
    int e = tile * 16 + (lane & 15);
    int sIdx = p.e_src[e], dIdx = p.e_dst[e];
    float dtv = p.e_dt[e];
    // gather xi -> cols [0,64), xj -> cols [64,128)  (f16, 16B vectors)
    #pragma unroll
    for (int i = 0; i < 8; ++i) {
      int idx = lane + 32 * i, row = idx >> 4, q = idx & 15;
      int dN = __shfl(dIdx, row, 32), sN = __shfl(sIdx, row, 32);
      int node = (q < 8) ? dN : sN;
      int off = (q & 7) * 8;
      *(float4*)&mW[row * 200 + ((q < 8) ? 0 : 64) + off] =
          *(const float4*)&p.x_h[(long)node * IN_F + off];
    }
    // time features: rel = cos(dt*freq + phase) -> cols [128,192)
    {
      int r = lane & 15, f0 = (lane >> 4) * 32;
      float dd = __shfl(dtv, r, 32);
      for (int f = f0; f < f0 + 32; ++f) {
        float c = __cosf(dd * p.freq[f] + p.phase[f]);
        mW[r * 200 + 128 + f] = (_Float16)c;
      }
    }
    asm volatile("" ::: "memory");  // wave-local LDS: same-wave DS ops are in-order
    int dNode[8];
    #pragma unroll
    for (int v = 0; v < 8; ++v) dNode[v] = __shfl(dIdx, v + ((lane >> 4) << 3), 32);
    // GEMM1: 16x96 = m(16x192) x W1t ; K chunks=6, N tiles=6
    Frag a[6];
    {
      int ar = lane & 15, ak = (lane >> 4) * 8;
      #pragma unroll
      for (int k = 0; k < 6; ++k) {
        const _Float16* q0 = &mW[ar * 200 + k * 32 + ak];
        a[k].f[0] = *(const float4*)q0; a[k].f[1] = *(const float4*)(q0 + 16);
      }
    }
    const int bc = lane & 15, bk = (lane >> 4) * 8;
    #pragma unroll
    for (int n = 0; n < 6; ++n) {
      v8f acc = {};   // inline SRC2=0 for first WMMA; bias folded into epilogue
      #pragma unroll
      for (int k = 0; k < 6; ++k) {
        Frag b; const _Float16* q0 = &p.tW1t[(n * 16 + bc) * 192 + k * 32 + bk];
        b.f[0] = *(const float4*)q0; b.f[1] = *(const float4*)(q0 + 16);
        acc = __builtin_amdgcn_wmma_f32_16x16x32_f16(false, a[k].v, false, b.v, (short)0, acc, false, false);
      }
      float bb = p.tb1[n * 16 + bc];
      #pragma unroll
      for (int v = 0; v < 8; ++v) {
        int row = v + ((lane >> 4) << 3);
        float t = fmaxf(acc[v] + bb, 0.f);
        h1W[row * 104 + n * 16 + bc] = (_Float16)t;
      }
    }
    asm volatile("" ::: "memory");
    // GEMM2: 16x64 = h1(16x96) x W2t ; K chunks=3, N tiles=4 ; tanh + scatter
    Frag a2[3];
    {
      int ar = lane & 15, ak = (lane >> 4) * 8;
      #pragma unroll
      for (int k = 0; k < 3; ++k) {
        const _Float16* q0 = &h1W[ar * 104 + k * 32 + ak];
        a2[k].f[0] = *(const float4*)q0; a2[k].f[1] = *(const float4*)(q0 + 16);
      }
    }
    #pragma unroll
    for (int n = 0; n < 4; ++n) {
      v8f acc = {};
      #pragma unroll
      for (int k = 0; k < 3; ++k) {
        Frag b; const _Float16* q0 = &p.tW2t[(n * 16 + bc) * 96 + k * 32 + bk];
        b.f[0] = *(const float4*)q0; b.f[1] = *(const float4*)(q0 + 16);
        acc = __builtin_amdgcn_wmma_f32_16x16x32_f16(false, a2[k].v, false, b.v, (short)0, acc, false, false);
      }
      float bb = p.tb2[n * 16 + bc];
      #pragma unroll
      for (int v = 0; v < 8; ++v) {
        int dN = dNode[v];
        int col = n * 16 + bc;
        float pe = tanh_fast(acc[v] + bb);
        float xi = p.x[(long)dN * IN_F + col];
        unsafeAtomicAdd(&p.aggA[(long)dN * IN_F + col], (2.f * pe - 1.f) * xi);
      }
    }
    if (lane < 16) unsafeAtomicAdd(&p.cntA[dIdx], 1.f);
  }
}

// -------- K3: h = relu((x + aggA/cnt) @ projW + b) ; zero aggB/cntB --------
__global__ __launch_bounds__(128, 1) void k_proj(P p) {
  __shared__ _Float16 mS[WV][16 * 72];   // 64 + pad 8
  const int lane = threadIdx.x & 31, wid = threadIdx.x >> 5;
  _Float16* mW = mS[wid];
  int w = blockIdx.x * WV + wid, nw = gridDim.x * WV;
  for (int tile = w; tile < N_SUB / 16; tile += nw) {
    int node0 = tile * 16;
    {
      int r = lane & 15, k0 = (lane >> 4) * 32;
      int node = node0 + r;
      float inv = 1.f / fmaxf(p.cntA[node], 1.f);
      for (int k = k0; k < k0 + 32; ++k) {
        float v = p.x[(long)node * IN_F + k] + p.aggA[(long)node * IN_F + k] * inv;
        mW[r * 72 + k] = (_Float16)v;
      }
    }
    asm volatile("" ::: "memory");
    Frag a[2];
    {
      int ar = lane & 15, ak = (lane >> 4) * 8;
      #pragma unroll
      for (int k = 0; k < 2; ++k) {
        const _Float16* q0 = &mW[ar * 72 + k * 32 + ak];
        a[k].f[0] = *(const float4*)q0; a[k].f[1] = *(const float4*)(q0 + 16);
      }
    }
    const int bc = lane & 15, bk = (lane >> 4) * 8;
    #pragma unroll
    for (int n = 0; n < 8; ++n) {
      v8f acc = {};
      #pragma unroll
      for (int k = 0; k < 2; ++k) {
        Frag b; const _Float16* q0 = &p.projWt[(n * 16 + bc) * 64 + k * 32 + bk];
        b.f[0] = *(const float4*)q0; b.f[1] = *(const float4*)(q0 + 16);
        acc = __builtin_amdgcn_wmma_f32_16x16x32_f16(false, a[k].v, false, b.v, (short)0, acc, false, false);
      }
      float bb = p.projb[n * 16 + bc];
      #pragma unroll
      for (int v = 0; v < 8; ++v) {
        int row = v + ((lane >> 4) << 3);
        long o = (long)(node0 + row) * HID + n * 16 + bc;
        float t = fmaxf(acc[v] + bb, 0.f);
        p.h[o] = t; p.h_h[o] = (_Float16)t; p.aggB[o] = 0.f;
      }
    }
    if (lane < 16) p.cntB[node0 + lane] = 0.f;
  }
}

// -------- K4: SMP edge MLP (m=256 -> 128 -> 128), scatter into agg/cnt --------
__global__ __launch_bounds__(128, 1) void k_edge_smp(P p, const _Float16* __restrict__ W1t,
                                                    const _Float16* __restrict__ W2t,
                                                    const float* __restrict__ b1,
                                                    const float* __restrict__ b2,
                                                    float* __restrict__ agg,
                                                    float* __restrict__ cnt) {
  __shared__ _Float16 mS[WV][16 * 264];   // 256 + pad 8
  __shared__ _Float16 h1S[WV][16 * 136];  // 128 + pad 8
  const int lane = threadIdx.x & 31, wid = threadIdx.x >> 5;
  _Float16* mW = mS[wid]; _Float16* h1W = h1S[wid];
  int w = blockIdx.x * WV + wid, nw = gridDim.x * WV;
  for (int tile = w; tile < N_EDGES / 16; tile += nw) {
    int e = tile * 16 + (lane & 15);
    int sIdx = p.e_src[e], dIdx = p.e_dst[e];
    #pragma unroll
    for (int i = 0; i < 16; ++i) {
      int idx = lane + 32 * i, row = idx >> 5, q = idx & 31;
      int dN = __shfl(dIdx, row, 32), sN = __shfl(sIdx, row, 32);
      int node = (q < 16) ? dN : sN;
      int off = (q & 15) * 8;
      *(float4*)&mW[row * 264 + ((q < 16) ? 0 : 128) + off] =
          *(const float4*)&p.h_h[(long)node * HID + off];
    }
    asm volatile("" ::: "memory");
    int dNode[8];
    #pragma unroll
    for (int v = 0; v < 8; ++v) dNode[v] = __shfl(dIdx, v + ((lane >> 4) << 3), 32);
    // GEMM1: 16x128 = m(16x256) x W1t ; K=8 chunks, N=8 tiles
    Frag a[8];
    {
      int ar = lane & 15, ak = (lane >> 4) * 8;
      #pragma unroll
      for (int k = 0; k < 8; ++k) {
        const _Float16* q0 = &mW[ar * 264 + k * 32 + ak];
        a[k].f[0] = *(const float4*)q0; a[k].f[1] = *(const float4*)(q0 + 16);
      }
    }
    const int bc = lane & 15, bk = (lane >> 4) * 8;
    #pragma unroll
    for (int n = 0; n < 8; ++n) {
      v8f acc = {};
      #pragma unroll
      for (int k = 0; k < 8; ++k) {
        Frag b; const _Float16* q0 = &W1t[(n * 16 + bc) * 256 + k * 32 + bk];
        b.f[0] = *(const float4*)q0; b.f[1] = *(const float4*)(q0 + 16);
        acc = __builtin_amdgcn_wmma_f32_16x16x32_f16(false, a[k].v, false, b.v, (short)0, acc, false, false);
      }
      float bb = b1[n * 16 + bc];
      #pragma unroll
      for (int v = 0; v < 8; ++v) {
        int row = v + ((lane >> 4) << 3);
        float t = fmaxf(acc[v] + bb, 0.f);
        h1W[row * 136 + n * 16 + bc] = (_Float16)t;
      }
    }
    asm volatile("" ::: "memory");
    // GEMM2: 16x128 = h1(16x128) x W2t ; K=4 chunks, N=8 tiles ; tanh + scatter
    Frag a2[4];
    {
      int ar = lane & 15, ak = (lane >> 4) * 8;
      #pragma unroll
      for (int k = 0; k < 4; ++k) {
        const _Float16* q0 = &h1W[ar * 136 + k * 32 + ak];
        a2[k].f[0] = *(const float4*)q0; a2[k].f[1] = *(const float4*)(q0 + 16);
      }
    }
    #pragma unroll
    for (int n = 0; n < 8; ++n) {
      v8f acc = {};
      #pragma unroll
      for (int k = 0; k < 4; ++k) {
        Frag b; const _Float16* q0 = &W2t[(n * 16 + bc) * 128 + k * 32 + bk];
        b.f[0] = *(const float4*)q0; b.f[1] = *(const float4*)(q0 + 16);
        acc = __builtin_amdgcn_wmma_f32_16x16x32_f16(false, a2[k].v, false, b.v, (short)0, acc, false, false);
      }
      float bb = b2[n * 16 + bc];
      #pragma unroll
      for (int v = 0; v < 8; ++v) {
        int dN = dNode[v];
        int col = n * 16 + bc;
        float pe = tanh_fast(acc[v] + bb);
        float hi = p.h[(long)dN * HID + col];
        unsafeAtomicAdd(&agg[(long)dN * HID + col], (2.f * pe - 1.f) * hi);
      }
    }
    if (lane < 16) unsafeAtomicAdd(&cnt[dIdx], 1.f);
  }
}

// -------- K5: h = BN(relu(h + aggB/cnt)) ; re-zero aggB and the other cnt --------
__global__ void k_bnres(P p, const float* __restrict__ cntR, float* __restrict__ cntZ, int bnOff) {
  long t = blockIdx.x * (long)blockDim.x + threadIdx.x;
  long nt = (long)gridDim.x * blockDim.x;
  for (long i = t; i < (long)N_SUB * HID; i += nt) {
    int node = (int)(i >> 7), col = (int)(i & 127);
    float inv = 1.f / fmaxf(cntR[node], 1.f);
    float v = p.h[i] + p.aggB[i] * inv;
    v = fmaxf(v, 0.f);
    v = v * p.bnScale[bnOff + col] + p.bnShift[bnOff + col];
    p.h[i] = v; p.h_h[i] = (_Float16)v; p.aggB[i] = 0.f;
  }
  for (long i = t; i < N_SUB; i += nt) cntZ[i] = 0.f;
}

// -------- K6: classifier on first BATCH rows (bias folded into bn shift) --------
__global__ __launch_bounds__(128, 1) void k_clf(P p) {
  __shared__ _Float16 mS[WV][16 * 136];
  __shared__ _Float16 z1S[WV][16 * 136];
  __shared__ _Float16 z2S[WV][16 * 72];
  const int lane = threadIdx.x & 31, wid = threadIdx.x >> 5;
  _Float16* mW = mS[wid]; _Float16* z1W = z1S[wid]; _Float16* z2W = z2S[wid];
  int w = blockIdx.x * WV + wid, nw = gridDim.x * WV;
  for (int tile = w; tile < BATCH / 16; tile += nw) {
    int n0 = tile * 16;
    #pragma unroll
    for (int i = 0; i < 8; ++i) {
      int idx = lane + 32 * i, row = idx >> 4, q = idx & 15;
      *(float4*)&mW[row * 136 + q * 8] = *(const float4*)&p.h_h[(long)(n0 + row) * HID + q * 8];
    }
    asm volatile("" ::: "memory");
    const int bc = lane & 15, bk = (lane >> 4) * 8;
    // layer 1: 128 -> 128, bn1(+b1) + relu
    Frag a[4];
    {
      int ar = lane & 15, ak = (lane >> 4) * 8;
      #pragma unroll
      for (int k = 0; k < 4; ++k) {
        const _Float16* q0 = &mW[ar * 136 + k * 32 + ak];
        a[k].f[0] = *(const float4*)q0; a[k].f[1] = *(const float4*)(q0 + 16);
      }
    }
    #pragma unroll
    for (int n = 0; n < 8; ++n) {
      v8f acc = {};
      #pragma unroll
      for (int k = 0; k < 4; ++k) {
        Frag b; const _Float16* q0 = &p.cW1t[(n * 16 + bc) * 128 + k * 32 + bk];
        b.f[0] = *(const float4*)q0; b.f[1] = *(const float4*)(q0 + 16);
        acc = __builtin_amdgcn_wmma_f32_16x16x32_f16(false, a[k].v, false, b.v, (short)0, acc, false, false);
      }
      int col = n * 16 + bc;
      float sc = p.bnScale[256 + col], sh = p.bnShift[256 + col];
      #pragma unroll
      for (int v = 0; v < 8; ++v) {
        int row = v + ((lane >> 4) << 3);
        float t = fmaxf(fmaf(acc[v], sc, sh), 0.f);
        z1W[row * 136 + col] = (_Float16)t;
      }
    }
    asm volatile("" ::: "memory");
    // layer 2: 128 -> 64, bn2(+b2) + relu
    Frag a2[4];
    {
      int ar = lane & 15, ak = (lane >> 4) * 8;
      #pragma unroll
      for (int k = 0; k < 4; ++k) {
        const _Float16* q0 = &z1W[ar * 136 + k * 32 + ak];
        a2[k].f[0] = *(const float4*)q0; a2[k].f[1] = *(const float4*)(q0 + 16);
      }
    }
    #pragma unroll
    for (int n = 0; n < 4; ++n) {
      v8f acc = {};
      #pragma unroll
      for (int k = 0; k < 4; ++k) {
        Frag b; const _Float16* q0 = &p.cW2t[(n * 16 + bc) * 128 + k * 32 + bk];
        b.f[0] = *(const float4*)q0; b.f[1] = *(const float4*)(q0 + 16);
        acc = __builtin_amdgcn_wmma_f32_16x16x32_f16(false, a2[k].v, false, b.v, (short)0, acc, false, false);
      }
      int col = n * 16 + bc;
      float sc = p.bnScale[384 + col], sh = p.bnShift[384 + col];
      #pragma unroll
      for (int v = 0; v < 8; ++v) {
        int row = v + ((lane >> 4) << 3);
        float t = fmaxf(fmaf(acc[v], sc, sh), 0.f);
        z2W[row * 72 + col] = (_Float16)t;
      }
    }
    asm volatile("" ::: "memory");
    // layer 3: 64 -> 1 dot product per row
    if (lane < 16) {
      float s = p.clfb3[0];
      for (int k = 0; k < 64; ++k) s += (float)z2W[lane * 72 + k] * p.clfW3[k];
      p.out[n0 + lane] = s;
    }
  }
}

extern "C" void kernel_launch(void* const* d_in, const int* in_sizes, int n_in,
                              void* d_out, int out_size, void* d_ws, size_t ws_size,
                              hipStream_t stream) {
  (void)in_sizes; (void)n_in; (void)out_size; (void)ws_size;
  // flattening: top-level dict insertion order; params pytree (sorted keys)
  P p{};
  p.x_all    = (const float*)d_in[0];
  p.node_ids = (const int*)d_in[1];
  p.e_src    = (const int*)d_in[2];
  p.e_dst    = ((const int*)d_in[2]) + N_EDGES;
  p.e_dt     = (const float*)d_in[3];
  // d_in[4] = batch_size (==BATCH)
  p.clfW1 = (const float*)d_in[5];  p.clfW2 = (const float*)d_in[6];  p.clfW3 = (const float*)d_in[7];
  p.clfb1 = (const float*)d_in[8];  p.clfb2 = (const float*)d_in[9];  p.clfb3 = (const float*)d_in[10];
  p.bn1b = (const float*)d_in[11]; p.bn1g = (const float*)d_in[12]; p.bn1m = (const float*)d_in[13]; p.bn1v = (const float*)d_in[14];
  p.bn2b = (const float*)d_in[15]; p.bn2g = (const float*)d_in[16]; p.bn2m = (const float*)d_in[17]; p.bn2v = (const float*)d_in[18];
  p.freq = (const float*)d_in[19]; p.phase = (const float*)d_in[20];
  p.projW = (const float*)d_in[21]; p.projb = (const float*)d_in[22];
  p.s0W1 = (const float*)d_in[23]; p.s0W2 = (const float*)d_in[24]; p.s0b1 = (const float*)d_in[25]; p.s0b2 = (const float*)d_in[26];
  p.s0bnb = (const float*)d_in[27]; p.s0bng = (const float*)d_in[28]; p.s0bnm = (const float*)d_in[29]; p.s0bnv = (const float*)d_in[30];
  p.s1W1 = (const float*)d_in[31]; p.s1W2 = (const float*)d_in[32]; p.s1b1 = (const float*)d_in[33]; p.s1b2 = (const float*)d_in[34];
  p.s1bnb = (const float*)d_in[35]; p.s1bng = (const float*)d_in[36]; p.s1bnm = (const float*)d_in[37]; p.s1bnv = (const float*)d_in[38];
  p.tW1 = (const float*)d_in[39]; p.tW2 = (const float*)d_in[40]; p.tb1 = (const float*)d_in[41]; p.tb2 = (const float*)d_in[42];

  char* ws = (char*)d_ws;
  size_t o = 0;
  p.x    = (float*)(ws + o);      o += (size_t)N_SUB * IN_F * 4;   // 25.6 MB
  p.x_h  = (_Float16*)(ws + o);   o += (size_t)N_SUB * IN_F * 2;   // 12.8 MB
  p.h    = (float*)(ws + o);      o += (size_t)N_SUB * HID * 4;    // 51.2 MB
  p.h_h  = (_Float16*)(ws + o);   o += (size_t)N_SUB * HID * 2;    // 25.6 MB
  p.aggA = (float*)(ws + o);      o += (size_t)N_SUB * IN_F * 4;   // 25.6 MB
  p.aggB = (float*)(ws + o);      o += (size_t)N_SUB * HID * 4;    // 51.2 MB
  p.cntA = (float*)(ws + o);      o += (size_t)N_SUB * 4;
  p.cntB = (float*)(ws + o);      o += (size_t)N_SUB * 4;
  _Float16* wf = (_Float16*)(ws + o);
  p.tW1t   = wf;           // 96 x 192
  p.tW2t   = wf + 18432;   // 64 x 96
  p.projWt = wf + 24576;   // 128 x 64
  p.s0W1t  = wf + 32768;   // 128 x 256
  p.s0W2t  = wf + 65536;   // 128 x 128
  p.s1W1t  = wf + 81920;
  p.s1W2t  = wf + 114688;
  p.cW1t   = wf + 131072;  // 128 x 128
  p.cW2t   = wf + 147456;  // 64 x 128
  o += 155648 * 2;
  o = (o + 255) & ~(size_t)255;
  p.bnScale = (float*)(ws + o); o += 448 * 4;
  p.bnShift = (float*)(ws + o); o += 448 * 4;
  p.out = (float*)d_out;

  k_prep  <<<64, 256, 0, stream>>>(p);
  k_gather<<<512, 256, 0, stream>>>(p);
  k_edge1 <<<4096, 128, 0, stream>>>(p);
  k_proj  <<<1024, 128, 0, stream>>>(p);
  k_edge_smp<<<4096, 128, 0, stream>>>(p, p.s0W1t, p.s0W2t, p.s0b1, p.s0b2, p.aggB, p.cntB);
  k_bnres <<<512, 256, 0, stream>>>(p, p.cntB, p.cntA, 0);
  k_edge_smp<<<4096, 128, 0, stream>>>(p, p.s1W1t, p.s1W2t, p.s1b1, p.s1b2, p.aggB, p.cntA);
  k_bnres <<<512, 256, 0, stream>>>(p, p.cntA, p.cntB, 128);
  k_clf   <<<160, 128, 0, stream>>>(p);
}
